// SH_Attention_8229157339328
// MI455X (gfx1250) — matrix operand order repaired and checked by reading, not compiled, not run on hardware
//
#include <hip/hip_runtime.h>
#include <hip/hip_bf16.h>

#define BB 8
#define SS 2048
#define EE 512
#define KIN 512
#define NEGINF -1000000.0f

typedef __bf16 bf16_t;
typedef __attribute__((ext_vector_type(16))) __bf16 v16bf;
typedef __attribute__((ext_vector_type(8)))  __bf16 v8bf;
typedef __attribute__((ext_vector_type(8)))  float  v8f;

// ---- WMMA fragment helpers (ISA 7.12.2 layouts) -------------------------
// A 16x32 (16-bit): lane -> row M, element i -> K = (i<8?i:i+8) + (lane>=16?8:0)
//   => two contiguous 8-element chunks at +kbase and +16+kbase
// B 32x16 (16-bit): lane -> col N, element i -> K = (i<8?i:i+8) + (lane>=16?8:0)
//   => two contiguous 8-element chunks at +kbase and +16+kbase
__device__ __forceinline__ v16bf cat8(v8bf lo, v8bf hi) {
  return __builtin_shufflevector(lo, hi, 0,1,2,3,4,5,6,7,8,9,10,11,12,13,14,15);
}
__device__ __forceinline__ v16bf load_a_frag_bf(const bf16_t* base, int lane) {
  const int kbase = (lane >= 16) ? 8 : 0;
  return cat8(*(const v8bf*)(base + kbase), *(const v8bf*)(base + 16 + kbase));
}
__device__ __forceinline__ v16bf load_b_frag(const bf16_t* base, int lane) {
  const int kbase = (lane >= 16) ? 8 : 0;
  return cat8(*(const v8bf*)(base + kbase), *(const v8bf*)(base + 16 + kbase));
}
__device__ __forceinline__ v16bf cvt16(v8f lo, v8f hi) {
  v16bf r;
#pragma unroll
  for (int i = 0; i < 8; ++i) { r[i] = (bf16_t)lo[i]; r[i + 8] = (bf16_t)hi[i]; }
  return r;
}
__device__ __forceinline__ v16bf load_a_frag_f32(const float* base, int lane) {
  const int kbase = (lane >= 16) ? 8 : 0;
  return cvt16(*(const v8f*)(base + kbase), *(const v8f*)(base + 16 + kbase));
}
#define WMMA_BF16(a, b, c) \
  __builtin_amdgcn_wmma_f32_16x16x32_bf16(false, (a), false, (b), (short)0, (c), false, false)

// ---------------------------------------------------------------------------
// Transpose + cast weights: Wt[e][k] = (bf16) W[k][e]   (512x512, trivial)
// ---------------------------------------------------------------------------
__global__ __launch_bounds__(256) void wt_bf16_kernel(
    const float* __restrict__ W, bf16_t* __restrict__ Wt)
{
  const int idx = blockIdx.x * 256 + threadIdx.x;   // over KIN*EE
  const int k = idx >> 9;           // row of W
  const int e = idx & (EE - 1);     // col of W
  Wt[(size_t)e * KIN + k] = (bf16_t)W[idx];
}

// ---------------------------------------------------------------------------
// Projection: Out = cast_bf16( (X[M,KIN] @ W) * out_scale )
// Wave owns a 16x64 strip (4 n-tiles sharing one A fragment per K-step).
// transpose_out: store Vt[b][e][s] instead of row-major [b][s][e].
// ---------------------------------------------------------------------------
__global__ __launch_bounds__(256) void proj_bf16_kernel(
    const float* __restrict__ X, const bf16_t* __restrict__ Wt,
    bf16_t* __restrict__ Out, float out_scale, int transpose_out)
{
  const int lane = threadIdx.x & 31;
  const int wave = threadIdx.x >> 5;
  const int groupsN = EE / 64;                       // 8 groups of 4 n-tiles
  const int gt = blockIdx.x * 8 + wave;
  const int tm = (gt / groupsN) * 16;
  const int tn = (gt % groupsN) * 64;
  const int mn = lane & 15;
  const int rbase = (lane >= 16) ? 8 : 0;

  v8f acc[4] = {v8f{}, v8f{}, v8f{}, v8f{}};
  const float* xrow = X + (size_t)(tm + mn) * KIN;
  for (int k0 = 0; k0 < KIN; k0 += 32) {
    const v16bf a = load_a_frag_f32(xrow + k0, lane);
#pragma unroll
    for (int j = 0; j < 4; ++j) {
      const v16bf b = load_b_frag(Wt + (size_t)(tn + j * 16 + mn) * KIN + k0, lane);
      acc[j] = WMMA_BF16(a, b, acc[j]);
    }
  }
  if (transpose_out) {
    const int bidx = tm / SS, sl = tm % SS;
#pragma unroll
    for (int j = 0; j < 4; ++j)
#pragma unroll
      for (int r = 0; r < 8; ++r)
        Out[((size_t)bidx * EE + tn + j * 16 + mn) * SS + sl + rbase + r] =
            (bf16_t)(acc[j][r] * out_scale);
  } else {
#pragma unroll
    for (int j = 0; j < 4; ++j)
#pragma unroll
      for (int r = 0; r < 8; ++r)
        Out[(size_t)(tm + rbase + r) * EE + tn + j * 16 + mn] =
            (bf16_t)(acc[j][r] * out_scale);
  }
}

// ---------------------------------------------------------------------------
// Attention: block = one (batch, 16-query-row) strip over all 2048 keys.
// Dynamic LDS: 16x2048 f32 score/prob tile (128 KB) + 256 f32 reduction pad.
// ---------------------------------------------------------------------------
__global__ __launch_bounds__(256) void attn_kernel(
    const bf16_t* __restrict__ Qb, const bf16_t* __restrict__ Kb,
    const bf16_t* __restrict__ Vt, const int* __restrict__ mask,
    float* __restrict__ Z, float* __restrict__ P)
{
  extern __shared__ float smem[];
  float* sc  = smem;              // [16][SS]
  float* red = smem + 16 * SS;    // [256]

  const int lane = threadIdx.x & 31;
  const int wave = threadIdx.x >> 5;
  const int b  = blockIdx.x / (SS / 16);
  const int qt = blockIdx.x % (SS / 16);

  const bf16_t* Q    = Qb + ((size_t)b * SS + (size_t)qt * 16) * EE;
  const bf16_t* Km   = Kb + (size_t)b * SS * EE;
  const bf16_t* Vb   = Vt + (size_t)b * EE * SS;     // [EE][SS] transposed
  const int*    Mrow = mask + ((size_t)b * SS + (size_t)qt * 16) * SS;
  float*        Prow = P    + ((size_t)b * SS + (size_t)qt * 16) * SS;
  float*        Zrow = Z    + ((size_t)b * SS + (size_t)qt * 16) * EE;

  const int mn = lane & 15;
  const int rbase = (lane >= 16) ? 8 : 0;

  // ---- Phase A: scores = Q K^T, mask -> LDS.  Wave owns 16 key tiles. ----
  const bf16_t* qrow = Q + (size_t)mn * EE;
  for (int g = 0; g < 4; ++g) {
    const int kt = wave * 16 + g * 4;                // 4 key tiles this group
    v8f acc[4] = {v8f{}, v8f{}, v8f{}, v8f{}};
    for (int e0 = 0; e0 < EE; e0 += 32) {
      const v16bf a = load_a_frag_bf(qrow + e0, lane);
#pragma unroll
      for (int j = 0; j < 4; ++j) {
        // B = K^T tile: lane column = key row (kt+j)*16+mn, K-dim = e (contiguous)
        const v16bf bb = load_b_frag(Km + (size_t)((kt + j) * 16 + mn) * EE + e0, lane);
        acc[j] = WMMA_BF16(a, bb, acc[j]);
      }
    }
#pragma unroll
    for (int j = 0; j < 4; ++j)
#pragma unroll
      for (int r = 0; r < 8; ++r) {
        const int row = rbase + r;
        const int col = (kt + j) * 16 + mn;
        float s = acc[j][r];
        if (Mrow[(size_t)row * SS + col] == 0) s = NEGINF;
        sc[row * SS + col] = s;
      }
  }
  __syncthreads();

  // ---- Phase B: row softmax (16 threads per row). Masked entries are exactly
  // NEGINF, so the post-softmax mask-multiply == zeroing where s==NEGINF
  // (also handles fully-masked rows); avoids re-reading the 134 MB mask. ----
  const int row = threadIdx.x >> 4;
  const int sub = threadIdx.x & 15;
  float mx = -3.0e38f;
  for (int c = sub; c < SS; c += 16) mx = fmaxf(mx, sc[row * SS + c]);
  red[threadIdx.x] = mx;
  __syncthreads();
  float rowmax = red[row * 16];
#pragma unroll
  for (int j = 1; j < 16; ++j) rowmax = fmaxf(rowmax, red[row * 16 + j]);
  __syncthreads();
  float sm = 0.0f;
  for (int c = sub; c < SS; c += 16) sm += __expf(sc[row * SS + c] - rowmax);
  red[threadIdx.x] = sm;
  __syncthreads();
  float rowsum = 0.0f;
#pragma unroll
  for (int j = 0; j < 16; ++j) rowsum += red[row * 16 + j];
  const float inv = 1.0f / rowsum;
  for (int c = sub; c < SS; c += 16) {
    const float s = sc[row * SS + c];
    float p = (s == NEGINF) ? 0.0f : __expf(s - rowmax) * inv;
    sc[row * SS + c] = p;
    Prow[(size_t)row * SS + c] = p;
  }
  __syncthreads();

  // ---- Phase C: Z = P @ V.  Wave owns 4 n-tiles sharing one A fragment. ----
  const float* prow = sc + mn * SS;
  const int nt0 = wave * 4;
  {
    v8f acc[4] = {v8f{}, v8f{}, v8f{}, v8f{}};
    const int kbase = (lane >= 16) ? 8 : 0;
    for (int k0 = 0; k0 < SS; k0 += 32) {
      const v8f lo = *(const v8f*)(prow + k0 + kbase);
      const v8f hi = *(const v8f*)(prow + k0 + 16 + kbase);
      const v16bf a = cvt16(lo, hi);
#pragma unroll
      for (int j = 0; j < 4; ++j) {
        // B from Vt: lane column = e = (nt0+j)*16+mn, K-dim = key (contiguous)
        const v16bf bb =
            load_b_frag(Vb + (size_t)((nt0 + j) * 16 + mn) * SS + k0, lane);
        acc[j] = WMMA_BF16(a, bb, acc[j]);
      }
    }
#pragma unroll
    for (int j = 0; j < 4; ++j)
#pragma unroll
      for (int r = 0; r < 8; ++r)
        Zrow[(size_t)(rbase + r) * EE + (nt0 + j) * 16 + mn] = acc[j][r];
  }
}

// ---------------------------------------------------------------------------
extern "C" void kernel_launch(void* const* d_in, const int* in_sizes, int n_in,
                              void* d_out, int out_size, void* d_ws, size_t ws_size,
                              hipStream_t stream)
{
  const float* Xq = (const float*)d_in[0];
  const float* Xk = (const float*)d_in[1];
  const float* Xv = (const float*)d_in[2];
  const int*   mk = (const int*)d_in[3];
  const float* Wq = (const float*)d_in[4];
  const float* Wk = (const float*)d_in[5];
  const float* Wv = (const float*)d_in[6];

  float* Z = (float*)d_out;                         // [B,S,E] f32
  float* P = Z + (size_t)BB * SS * EE;              // [B,S,S] f32

  const size_t nQKV = (size_t)BB * SS * EE;         // 8.39M elems
  bf16_t* Qb  = (bf16_t*)d_ws;                      // [B][S][E] bf16
  bf16_t* Kb  = Qb + nQKV;                          // [B][S][E] bf16
  bf16_t* Vt  = Kb + nQKV;                          // [B][E][S] bf16 (transposed)
  bf16_t* WtQ = Vt + nQKV;                          // [E][K] bf16
  bf16_t* WtK = WtQ + (size_t)KIN * EE;
  bf16_t* WtV = WtK + (size_t)KIN * EE;

  const float inv_scale = 0.21022410381342865f;     // 1 / 512^(1/4)

  dim3 blk(256);
  const int wgrid = (KIN * EE) / 256;               // 1024
  wt_bf16_kernel<<<dim3(wgrid), blk, 0, stream>>>(Wq, WtQ);
  wt_bf16_kernel<<<dim3(wgrid), blk, 0, stream>>>(Wk, WtK);
  wt_bf16_kernel<<<dim3(wgrid), blk, 0, stream>>>(Wv, WtV);

  // 1024 m-tiles x 8 n-groups = 8192 wave-jobs, 8 waves/block -> 1024 blocks
  const int pgrid = (BB * SS / 16) * (EE / 64) / 8;
  proj_bf16_kernel<<<dim3(pgrid), blk, 0, stream>>>(Xq, WtQ, Qb, inv_scale, 0);
  proj_bf16_kernel<<<dim3(pgrid), blk, 0, stream>>>(Xk, WtK, Kb, inv_scale, 0);
  proj_bf16_kernel<<<dim3(pgrid), blk, 0, stream>>>(Xv, WtV, Vt, 1.0f, 1);

  const size_t smem = (size_t)(16 * SS + 256) * sizeof(float);  // ~132 KB of 320 KB WGP LDS
  attn_kernel<<<dim3(BB * SS / 16), blk, smem, stream>>>(Qb, Kb, Vt, mk, Z, P);
}
